// SingleWindowAttentionKAN_11321533792852
// MI455X (gfx1250) — matrix operand order repaired and checked by reading, not compiled, not run on hardware
//
#include <hip/hip_runtime.h>
#include <stdint.h>

// ---------------- CDNA5 WMMA types ----------------
typedef __attribute__((ext_vector_type(16))) __bf16 v16bf;
typedef __attribute__((ext_vector_type(8)))  float  v8f;

#define DIM    256
#define NTOK   49           // tokens per window
#define HEADS  8
#define HD     32           // head dim = one bf16 WMMA K-step
#define BATCH  2048
#define TOK    (BATCH*NTOK) // 100352 tokens
#define KFEAT  2304         // 256 silu + 256*8 spline features
#define TILE_M 64
#define NBLK   (TOK/TILE_M) // 1568 (exact)
#define KSTEPS (KFEAT/32)   // 72

// ---- bf16 bit-level helpers (RNE) ----
__device__ __forceinline__ unsigned short f2bf(float f) {
  union { float f; unsigned u; } v; v.f = f;
  unsigned u = v.u;
  u = (u + 0x7FFFu + ((u >> 16) & 1u)) >> 16;
  return (unsigned short)u;
}
__device__ __forceinline__ unsigned pack2(float a, float b) {
  return (unsigned)f2bf(a) | ((unsigned)f2bf(b) << 16);
}
__device__ __forceinline__ __bf16 bfbits(unsigned short u) {
  __bf16 r; __builtin_memcpy(&r, &u, 2); return r;
}
__device__ __forceinline__ v8f vzero8() {
  v8f z = {0.f,0.f,0.f,0.f,0.f,0.f,0.f,0.f}; return z;
}

// ---- efficient-KAN B-spline bases: grid t[j] = (j-3)*0.4 - 1.0, j in [0,11]; order 3 ----
__device__ __forceinline__ void bspline8(float x, float bs[8]) {
  const float h = 0.4f;
  float t[12];
#pragma unroll
  for (int j = 0; j < 12; ++j) t[j] = (float)(j - 3) * h - 1.0f;
  float b[11];
#pragma unroll
  for (int j = 0; j < 11; ++j) b[j] = (x >= t[j] && x < t[j + 1]) ? 1.0f : 0.0f;
#pragma unroll
  for (int p = 1; p <= 3; ++p) {
    const float inv = 1.0f / ((float)p * h);   // uniform grid: t[j+p]-t[j] == p*h
#pragma unroll
    for (int j = 0; j < 10; ++j) {
      if (j < 11 - p)
        b[j] = (x - t[j]) * inv * b[j] + (t[j + p + 1] - x) * inv * b[j + 1];
    }
  }
#pragma unroll
  for (int j = 0; j < 8; ++j) bs[j] = b[j];
}

// B-fragment swizzle for a (k, n) weight element within a 32xN slab:
//   lane = (n&15) + ((k>>3)&1 ? 16 : 0), e = (k&7) + (k&16 ? 8 : 0); tiles: [kstep][ntile][lane][e]

// ---------------- Kernel 1: weight prep (f32 -> bf16, pre-swizzled to B-fragment order) ----------------
__global__ __launch_bounds__(256) void prep_weights_kernel(
    const float* __restrict__ qb, const float* __restrict__ qs,
    const float* __restrict__ kb, const float* __restrict__ ks,
    const float* __restrict__ vb, const float* __restrict__ vs,
    const float* __restrict__ pw,
    unsigned short* __restrict__ Wall, unsigned short* __restrict__ Wp)
{
  const int gid = blockIdx.x * 256 + threadIdx.x;
  const int WSZ = KFEAT * DIM; // 589824
  if (gid < 3 * WSZ) {
    const int m = gid / WSZ;
    const int rem = gid - m * WSZ;
    const int r = rem >> 8, o = rem & 255;   // r = feature row (K), o = output col (N)
    const float* base = (m == 0) ? qb : (m == 1) ? kb : vb;
    const float* spl  = (m == 0) ? qs : (m == 1) ? ks : vs;
    const float v = (r < DIM) ? base[rem] : spl[rem - DIM * DIM];
    const int krel = r & 31;
    const int dst = (((r >> 5) * 16 + (o >> 4)) * 32 + ((o & 15) + (((krel >> 3) & 1) ? 16 : 0))) * 16 + (krel & 7) + ((krel & 16) ? 8 : 0);
    Wall[(size_t)m * WSZ + dst] = f2bf(v);
  } else {
    const int idx = gid - 3 * WSZ;
    if (idx < DIM * DIM) {
      const int r = idx >> 8, o = idx & 255;
      const int krel = r & 31;
      const int dst = (((r >> 5) * 16 + (o >> 4)) * 32 + ((o & 15) + (((krel >> 3) & 1) ? 16 : 0))) * 16 + (krel & 7) + ((krel & 16) ? 8 : 0);
      Wp[dst] = f2bf(pw[idx]);
    }
  }
}

// ---------------- Kernel 2: fused KAN features + GEMM (tokens x 2304) @ (2304 x 256) ----------------
// Features are written directly into the 16-bit A-fragment layout in LDS (288 KB of the 320 KB WGP LDS);
// B fragments are contiguous 32B/lane global loads from the pre-swizzled weights (WGP$/L2 resident).
__global__ __launch_bounds__(256) void kan_qkv_kernel(
    const float* __restrict__ x,
    const unsigned short* __restrict__ Wall,
    unsigned short* __restrict__ qout,
    unsigned short* __restrict__ kout,
    unsigned short* __restrict__ vout)
{
  extern __shared__ unsigned short smem[];
  unsigned short* ldsA = smem;            // 72*4*32*16 = 147456 ushorts (288 KB)

  const int tid  = threadIdx.x;
  const int proj = blockIdx.y;
  const unsigned short* W = Wall + (size_t)proj * (KFEAT * DIM);
  unsigned short* out = (proj == 0) ? qout : (proj == 1) ? kout : vout;
  const int tok0 = blockIdx.x * TILE_M;

  // ---- feature generation: each job = 4 input dims of one token (float4 load)
  const float4* xv4 = (const float4*)(x + (size_t)tok0 * DIM);
  for (int i = 0; i < 16; ++i) {
    const int j = i * 256 + tid;           // 0..4095
    const int token = j >> 6;              // 0..63
    const int d0 = (j & 63) * 4;           // 0,4,...,252
    const float4 xv = xv4[j];
    const int tileRow = token >> 4;
    {
      // silu branch -> feature rows k = d0..d0+3 (contiguous in A-frag layout)
      const float s0 = xv.x / (1.0f + __expf(-xv.x));
      const float s1 = xv.y / (1.0f + __expf(-xv.y));
      const float s2 = xv.z / (1.0f + __expf(-xv.z));
      const float s3 = xv.w / (1.0f + __expf(-xv.w));
      const int krel = d0 & 31;
      const int hi = (krel >> 3) & 1;
      const int e0 = (krel & 7) + ((krel & 16) ? 8 : 0);
      uint2 p; p.x = pack2(s0, s1); p.y = pack2(s2, s3);
      *(uint2*)(ldsA + ((((d0 >> 5) * 4 + tileRow) * 32 + ((token & 15) + hi * 16)) * 16 + e0)) = p;
    }
    const float xd[4] = {xv.x, xv.y, xv.z, xv.w};
#pragma unroll
    for (int dd = 0; dd < 4; ++dd) {
      const int d = d0 + dd;
      float bsv[8]; bspline8(xd[dd], bsv);
      // spline branch -> feature rows k0 = 256 + d*8 .. +7 (one contiguous b128 in A-frag layout)
      const int k0 = DIM + d * 8;
      const int krel = k0 & 31;            // (d%4)*8
      const int hi = (krel >> 3) & 1;
      const int e0 = (krel & 16) ? 8 : 0;
      uint4 p;
      p.x = pack2(bsv[0], bsv[1]); p.y = pack2(bsv[2], bsv[3]);
      p.z = pack2(bsv[4], bsv[5]); p.w = pack2(bsv[6], bsv[7]);
      *(uint4*)(ldsA + ((((k0 >> 5) * 4 + tileRow) * 32 + ((token & 15) + hi * 16)) * 16 + e0)) = p;
    }
  }
  __syncthreads();

  const int lane = tid & 31, wave = tid >> 5;
  const int wr = wave >> 1, wc = wave & 1;   // 4x2 wave grid: 16 tokens x 128 cols per wave

  v8f acc[8];
#pragma unroll
  for (int i = 0; i < 8; ++i) acc[i] = vzero8();

  for (int ksi = 0; ksi < KSTEPS; ++ksi) {
    const v16bf a = *(const v16bf*)(ldsA + ((ksi * 4 + wr) * 32 + lane) * 16);
    const unsigned short* Wb = W + (size_t)ksi * 8192 + (size_t)(wc * 8) * 512 + (size_t)lane * 16;
    v16bf bm[8];
#pragma unroll
    for (int nt = 0; nt < 8; ++nt)
      bm[nt] = *(const v16bf*)(Wb + nt * 512);
#pragma unroll
    for (int nt = 0; nt < 8; ++nt)
      acc[nt] = __builtin_amdgcn_wmma_f32_16x16x32_bf16(false, a, false, bm[nt], (short)0, acc[nt], false, false);
  }

  const int hi = lane >> 4, lo = lane & 15;
#pragma unroll
  for (int nt = 0; nt < 8; ++nt) {
    const int col = (wc * 8 + nt) * 16 + lo;
#pragma unroll
    for (int r = 0; r < 8; ++r) {
      const int row = tok0 + wr * 16 + r + hi * 8;   // C layout: M = r + hi*8
      out[(size_t)row * DIM + col] = f2bf(acc[nt][r]);
    }
  }
}

// ---------------- Kernel 3: windowed attention per (window, head) ----------------
__global__ __launch_bounds__(128) void attn_kernel(
    const unsigned short* __restrict__ qg,
    const unsigned short* __restrict__ kg,
    const unsigned short* __restrict__ vg,
    const float* __restrict__ bias_table,
    const int* __restrict__ rel_index,
    unsigned short* __restrict__ ctx)
{
  __shared__ unsigned short qs[4 * 32 * 16];      // A-frag order: [rowTile][lane][e]
  __shared__ unsigned short ksm[4 * 32 * 16];     // B-frag order: [colTile][lane][e]
  __shared__ unsigned short vsm[2 * 2 * 32 * 16]; // B-frag order: [s2][ct2][lane][e]
  __shared__ unsigned short Pst[4 * 2 * 32 * 16]; // A-frag order: [wave][s2][lane][e]
  __shared__ float biasL[NTOK * NTOK];

  const int b = blockIdx.x, h = blockIdx.y;
  const int tid = threadIdx.x, lane = tid & 31, w = tid >> 5;
  const int hi = lane >> 4, lo = lane & 15;

  // stage q/k head slices (zero-pad to 64 rows), swizzled into fragment order
  for (int j = tid; j < 256; j += 128) {          // 256 uint4 jobs per matrix
    const int m = j >> 2, d0 = (j & 3) * 8;
    uint4 qv = {0u,0u,0u,0u}, kv = {0u,0u,0u,0u};
    if (m < NTOK) {
      const size_t base = ((size_t)(b * NTOK + m)) * DIM + h * HD;
      qv = ((const uint4*)(qg + base))[j & 3];
      kv = ((const uint4*)(kg + base))[j & 3];
    }
    const int hiq = (d0 >> 3) & 1;
    const int e0q = (d0 & 16) ? 8 : 0;
    *(uint4*)(qs + (((m >> 4) * 32 + (m & 15) + hiq * 16) * 16 + e0q)) = qv;
    const int hik = ((d0 >> 3) & 1) ? 16 : 0;
    *(uint4*)(ksm + (((m >> 4) * 32 + (m & 15) + hik) * 16 + ((d0 & 16) ? 8 : 0))) = kv;
  }
  // stage v column-wise: thread = (dim, 8-row group); 8 K-rows pack into one b128
#pragma unroll
  for (int g2 = 0; g2 < 2; ++g2) {
    const int d = tid & 31;
    const int r0 = ((tid >> 5) + g2 * 4) * 8;     // 0,8,...,56
    unsigned short tmp[8];
#pragma unroll
    for (int rr = 0; rr < 8; ++rr) {
      const int row = r0 + rr;
      tmp[rr] = (row < NTOK) ? vg[((size_t)(b * NTOK + row)) * DIM + h * HD + d] : (unsigned short)0;
    }
    const int s2 = r0 >> 5;
    const int hiv = ((r0 & 31) >= 16) ? 1 : 0;
    uint4 p;
    p.x = (unsigned)tmp[0] | ((unsigned)tmp[1] << 16);
    p.y = (unsigned)tmp[2] | ((unsigned)tmp[3] << 16);
    p.z = (unsigned)tmp[4] | ((unsigned)tmp[5] << 16);
    p.w = (unsigned)tmp[6] | ((unsigned)tmp[7] << 16);
    *(uint4*)(vsm + (((s2 * 2 + (d >> 4)) * 32 + (d & 15) + hiv * 16) * 16 + (r0 & 15))) = p;
  }
  for (int i = tid; i < NTOK * NTOK; i += 128)
    biasL[i] = bias_table[rel_index[i] * HEADS + h];
  __syncthreads();

  // S = Q * K^T (K = head dim = one WMMA K-step)
  const v16bf qa = *(const v16bf*)(qs + (w * 32 + lane) * 16);
  v8f sacc[4];
#pragma unroll
  for (int ct = 0; ct < 4; ++ct) {
    const v16bf kf = *(const v16bf*)(ksm + (ct * 32 + lane) * 16);
    v8f z = vzero8();
    sacc[ct] = __builtin_amdgcn_wmma_f32_16x16x32_bf16(false, qa, false, kf, (short)0, z, false, false);
  }

  const float scale = 0.17677669529663687f;       // 1/sqrt(32)
#pragma unroll
  for (int ct = 0; ct < 4; ++ct) {
#pragma unroll
    for (int r = 0; r < 8; ++r) {
      const int qi = w * 16 + r + hi * 8, kj = ct * 16 + lo;
      float s = sacc[ct][r] * scale;
      s += (qi < NTOK && kj < NTOK) ? biasL[qi * NTOK + kj] : 0.0f;
      s = (kj < NTOK) ? s : -1e30f;               // mask padded key columns
      s = (qi < NTOK) ? s : 0.0f;                 // keep padded query rows finite
      sacc[ct][r] = s;
    }
  }

  // softmax: rows live in (r, hi); cols across lo lanes (xor<16 stays in hi group) + 4 tiles
#pragma unroll
  for (int r = 0; r < 8; ++r) {
    float m = -1e30f;
#pragma unroll
    for (int ct = 0; ct < 4; ++ct) m = fmaxf(m, sacc[ct][r]);
#pragma unroll
    for (int off = 8; off >= 1; off >>= 1) m = fmaxf(m, __shfl_xor(m, off, 32));
    float p[4], sum = 0.0f;
#pragma unroll
    for (int ct = 0; ct < 4; ++ct) { p[ct] = __expf(sacc[ct][r] - m); sum += p[ct]; }
#pragma unroll
    for (int off = 8; off >= 1; off >>= 1) sum += __shfl_xor(sum, off, 32);
    const float inv = 1.0f / sum;
    // write P directly in A-fragment order: row rr = r+hi*8, col kj = ct*16+lo
#pragma unroll
    for (int ct = 0; ct < 4; ++ct) {
      const int kj = ct * 16 + lo;
      const int krel = kj & 31;
      const int ahi = (krel >> 4) & 1;
      const int ae = krel & 15;
      Pst[((w * 2 + (kj >> 5)) * 32 + (r + hi * 8) + ahi * 16) * 16 + ae] = f2bf(p[ct] * inv);
    }
  }
  __syncthreads();

  // ctx = P(49x64pad) * V(64x32): 2 K-steps, 2 dim tiles
  v8f cacc[2] = { vzero8(), vzero8() };
#pragma unroll
  for (int s2 = 0; s2 < 2; ++s2) {
    const v16bf pa = *(const v16bf*)(Pst + ((w * 2 + s2) * 32 + lane) * 16);
#pragma unroll
    for (int ct2 = 0; ct2 < 2; ++ct2) {
      const v16bf vf = *(const v16bf*)(vsm + ((s2 * 2 + ct2) * 32 + lane) * 16);
      cacc[ct2] = __builtin_amdgcn_wmma_f32_16x16x32_bf16(false, pa, false, vf, (short)0, cacc[ct2], false, false);
    }
  }

#pragma unroll
  for (int ct2 = 0; ct2 < 2; ++ct2) {
#pragma unroll
    for (int r = 0; r < 8; ++r) {
      const int q = w * 16 + r + hi * 8;
      if (q < NTOK)
        ctx[((size_t)(b * NTOK + q)) * DIM + h * HD + ct2 * 16 + lo] = f2bf(cacc[ct2][r]);
    }
  }
}

// ---------------- Kernel 4: output projection (tokens x 256) @ (256 x 256) + bias ----------------
__global__ __launch_bounds__(256) void proj_kernel(
    const unsigned short* __restrict__ ctx,
    const unsigned short* __restrict__ Wp,   // pre-swizzled B-fragment order
    const float* __restrict__ pb,
    float* __restrict__ out)
{
  extern __shared__ unsigned short sm2[];
  unsigned short* cL = sm2;                 // ctx tile in A-fragment order: [ks][rowTile][lane][e]

  const int tid = threadIdx.x;
  const int tok0 = blockIdx.x * TILE_M;
  for (int j = tid; j < 2048; j += 256) {   // 2048 uint4 = 64 rows x 256 cols
    const int m = j >> 5, k0 = (j & 31) * 8;
    const uint4 cv = *(const uint4*)(ctx + (size_t)(tok0 + m) * DIM + k0);
    const int krel = k0 & 31;               // 0,8,16,24
    const int ahi = (krel >> 3) & 1;
    const int e0 = (krel & 16) ? 8 : 0;
    *(uint4*)(cL + ((((k0 >> 5) * 4 + (m >> 4)) * 32 + (m & 15) + ahi * 16) * 16 + e0)) = cv;
  }
  __syncthreads();

  const int lane = tid & 31, wave = tid >> 5;
  const int wr = wave >> 1, wc = wave & 1;
  const int hi = lane >> 4, lo = lane & 15;

  v8f acc[8];
#pragma unroll
  for (int i = 0; i < 8; ++i) acc[i] = vzero8();

#pragma unroll
  for (int ksi = 0; ksi < 8; ++ksi) {
    const v16bf a = *(const v16bf*)(cL + ((ksi * 4 + wr) * 32 + lane) * 16);
    const unsigned short* Wb = Wp + (size_t)(ksi * 16 + wc * 8) * 512 + (size_t)lane * 16;
    v16bf bm[8];
#pragma unroll
    for (int nt = 0; nt < 8; ++nt)
      bm[nt] = *(const v16bf*)(Wb + nt * 512);
#pragma unroll
    for (int nt = 0; nt < 8; ++nt)
      acc[nt] = __builtin_amdgcn_wmma_f32_16x16x32_bf16(false, a, false, bm[nt], (short)0, acc[nt], false, false);
  }

#pragma unroll
  for (int nt = 0; nt < 8; ++nt) {
    const int col = (wc * 8 + nt) * 16 + lo;
    const float bias = pb[col];
#pragma unroll
    for (int r = 0; r < 8; ++r) {
      const int row = tok0 + wr * 16 + r + hi * 8;
      out[(size_t)row * DIM + col] = acc[nt][r] + bias;
    }
  }
}

// ---------------- launch ----------------
extern "C" void kernel_launch(void* const* d_in, const int* in_sizes, int n_in,
                              void* d_out, int out_size, void* d_ws, size_t ws_size,
                              hipStream_t stream)
{
  const float* x        = (const float*)d_in[0];
  const float* q_base   = (const float*)d_in[1];
  const float* q_spline = (const float*)d_in[2];
  const float* k_base   = (const float*)d_in[3];
  const float* k_spline = (const float*)d_in[4];
  const float* v_base   = (const float*)d_in[5];
  const float* v_spline = (const float*)d_in[6];
  const float* proj_w   = (const float*)d_in[7];
  const float* proj_b   = (const float*)d_in[8];
  const float* bias_tab = (const float*)d_in[9];
  const int*   rel_idx  = (const int*)d_in[10];
  float* out = (float*)d_out;

  unsigned short* ws = (unsigned short*)d_ws;
  const size_t WSZ = (size_t)KFEAT * DIM;          // 589824 elems per projection weight
  unsigned short* Wall = ws;                        // 3*WSZ bf16 (B-frag order)
  unsigned short* Wp   = Wall + 3 * WSZ;            // 65536 bf16 (B-frag order)
  unsigned short* qbuf = Wp + (size_t)DIM * DIM;    // TOK*DIM bf16
  unsigned short* kbuf = qbuf + (size_t)TOK * DIM;
  unsigned short* vbuf = kbuf + (size_t)TOK * DIM;
  unsigned short* ctxb = qbuf;                      // alias: q dead after attention; per-block
                                                    // region is read -> barrier -> written

  prep_weights_kernel<<<7168, 256, 0, stream>>>(q_base, q_spline, k_base, k_spline,
                                                v_base, v_spline, proj_w, Wall, Wp);
  kan_qkv_kernel<<<dim3(NBLK, 3), 256, 147456 * sizeof(unsigned short), stream>>>(
      x, Wall, qbuf, kbuf, vbuf);
  attn_kernel<<<dim3(BATCH, HEADS), 128, 0, stream>>>(qbuf, kbuf, vbuf, bias_tab, rel_idx, ctxb);
  proj_kernel<<<NBLK, 256, 16384 * sizeof(unsigned short), stream>>>(ctxb, Wp, proj_b, out);
}
